// DynamicInteraction_49331994362264
// MI455X (gfx1250) — matrix operand order repaired and checked by reading, not compiled, not run on hardware
//
#include <hip/hip_runtime.h>

// ------------------------ problem constants ------------------------
#define B_    8
#define C_    64
#define N_    4096
#define S_    32
#define CR_   16
#define O1_   2048        // 2*C*R*C
#define NT_   16          // n-columns per workgroup
#define EPS_  1e-5f

// ------------------------ vector types ------------------------
typedef __attribute__((ext_vector_type(16))) __bf16         v16bf;
typedef __attribute__((ext_vector_type(2)))  __bf16         v2bf;
typedef __attribute__((ext_vector_type(8)))  float          v8f;
typedef __attribute__((ext_vector_type(4)))  float          f32x4;
typedef __attribute__((ext_vector_type(16))) unsigned short u16x16;
typedef __attribute__((ext_vector_type(8)))  unsigned short u16x8;
typedef __attribute__((ext_vector_type(8)))  unsigned int   u32x8;

// ------------------------ LDS layout (bytes) -----------------------
#define OFF_FB    0                     // features^T tile [16 n][64 c] bf16     2 KB
#define OFF_P1T   2048                  // param1^T [16 n][16 r][64 c] bf16     32 KB
#define OFF_P2T   (OFF_P1T + 32768)     // param2^T [16 n][64 c][16 r] bf16     32 KB
#define OFF_FLATT (OFF_P2T + 32768)     // flat^T   [16 n][2048 k]     bf16     64 KB
#define OFF_YACC  (OFF_FLATT + 65536)   // y accum  [64 o][16 n]       f32       4 KB
#define OFF_SCR   (OFF_YACC + 4096)     // per-wave scratch: 8 x 12 KB          96 KB
#define SCR_BYTES 12288
#define LDS_TOTAL (OFF_SCR + 8 * SCR_BYTES)   // 235,520 B < 320 KB WGP LDS

#define RSQ(x) __builtin_amdgcn_rsqf(x)   // raw v_rsq_f32, no denorm fixup

// native bf16 converts (hardware v_cvt_*_bf16_f32)
static __device__ __forceinline__ unsigned short f2bf(float x) {
  return __builtin_bit_cast(unsigned short, (__bf16)x);
}
static __device__ __forceinline__ unsigned int pk2(float a, float b) {
  v2bf v; v[0] = (__bf16)a; v[1] = (__bf16)b;
  return __builtin_bit_cast(unsigned int, v);
}

static __device__ __forceinline__ v8f wmma_bf16(v16bf a, v16bf b, v8f c) {
  // (neg_a, A, neg_b, B, c_mod, C, reuse_a, reuse_b)
  return __builtin_amdgcn_wmma_f32_16x16x32_bf16(false, a, false, b, (short)0, c,
                                                 false, false);
}

// A fragment from LDS bf16, row-major [M][K]: elems 0..7 at p, 8..15 at p+16
static __device__ __forceinline__ v16bf a_from_lds(const unsigned short* p) {
  u16x8 lo = *(const u16x8*)p;
  u16x8 hi = *(const u16x8*)(p + 16);
  u16x16 t;
#pragma unroll
  for (int e = 0; e < 8; ++e) { t[e] = lo[e]; t[8 + e] = hi[e]; }
  return __builtin_bit_cast(v16bf, t);
}

// B fragment from LDS bf16, B^T row-major [N][K]: elems 0..7 at p, 8..15 at p+16
static __device__ __forceinline__ v16bf b_from_lds(const unsigned short* p) {
  u16x8 b0 = *(const u16x8*)p;
  u16x8 b1 = *(const u16x8*)(p + 16);
  u16x16 t;
#pragma unroll
  for (int e = 0; e < 8; ++e) { t[e] = b0[e]; t[8 + e] = b1[e]; }
  return __builtin_bit_cast(v16bf, t);
}

// A fragment from global fp32 row (cvt to bf16): 8 floats at p, 8 at p+16
static __device__ __forceinline__ v16bf a_from_global(const float* p) {
  const f32x4* q = (const f32x4*)p;
  f32x4 x0 = q[0], x1 = q[1], x2 = q[4], x3 = q[5];
  u32x8 t;
  t[0] = pk2(x0[0], x0[1]); t[1] = pk2(x0[2], x0[3]);
  t[2] = pk2(x1[0], x1[1]); t[3] = pk2(x1[2], x1[3]);
  t[4] = pk2(x2[0], x2[1]); t[5] = pk2(x2[2], x2[3]);
  t[6] = pk2(x3[0], x3[1]); t[7] = pk2(x3[2], x3[3]);
  return __builtin_bit_cast(v16bf, t);
}

#define WB() __builtin_amdgcn_wave_barrier()

__global__ __launch_bounds__(256)
void dyninter_fused(const float* __restrict__ pre,   // [B,C,N,S]
                    const float* __restrict__ feat,  // [B,C,N]
                    const float* __restrict__ w1,    // [O1,C]
                    const float* __restrict__ bn1_g, const float* __restrict__ bn1_b,
                    const float* __restrict__ bn1_m, const float* __restrict__ bn1_v,
                    const float* __restrict__ ln1_g, const float* __restrict__ ln1_b,
                    const float* __restrict__ ln2_g, const float* __restrict__ ln2_b,
                    const float* __restrict__ w2,    // [C, C*S]
                    const float* __restrict__ bn2_g, const float* __restrict__ bn2_b,
                    const float* __restrict__ bn2_m, const float* __restrict__ bn2_v,
                    float* __restrict__ out) {       // [B,C,N]
  extern __shared__ __align__(16) char smem[];
  unsigned short* fB    = (unsigned short*)(smem + OFF_FB);
  unsigned short* p1T   = (unsigned short*)(smem + OFF_P1T);
  unsigned short* p2T   = (unsigned short*)(smem + OFF_P2T);
  unsigned short* flatT = (unsigned short*)(smem + OFF_FLATT);
  float*          yacc  = (float*)(smem + OFF_YACC);

  const int tid  = threadIdx.x;
  const int lane = tid & 31;
  const int wv   = tid >> 5;
  const int half = lane >> 4;   // which 16-lane half
  const int l16  = lane & 15;

  char* scr = smem + OFF_SCR + wv * SCR_BYTES;
  unsigned short* preA  = (unsigned short*)scr;          // [32 s][64 c] bf16, 4 KB
  float*          t1raw = (float*)(scr + 4096);          // [32 s][16 r] f32,  2 KB
  unsigned short* t1A   = (unsigned short*)(scr + 6144); // [32 s][16 r] bf16, 1 KB
  float*          t2raw = (float*)(scr + 4096);          // [32 s][64 c] f32,  8 KB (reuse)

  const int blk = blockIdx.x;       // 0 .. B_*(N_/NT_)-1
  const int b   = blk >> 8;         // N_/NT_ == 256
  const int n0  = (blk & 255) * NT_;

  // ---------------- phase 0: stage features tile, zero y accumulator --------
#pragma unroll
  for (int q = 0; q < 4; ++q) yacc[tid + q * 256] = 0.f;
#pragma unroll
  for (int q = 0; q < 4; ++q) {
    const int idx = tid + q * 256;          // 0..1023
    const int c = idx >> 4, j = idx & 15;
    fB[j * C_ + c] = f2bf(feat[((size_t)(b * C_ + c)) * N_ + n0 + j]);
  }
  __syncthreads();

  // ---------------- phase 1: dyn = relu(BN1(w1 @ features)) -> p1T/p2T ------
  // GEMM M=2048, N=16, K=64 : 128 M-tiles, 16 per wave, 2 k-steps each.
#pragma unroll 1
  for (int i = 0; i < 16; ++i) {
    const int mt  = i * 8 + wv;              // 0..127 (uniform per wave)
    const int row = mt * 16 + l16;           // o-row for this lane's A data
    v8f acc = {};
#pragma unroll
    for (int ks = 0; ks < 2; ++ks) {
      v16bf a  = a_from_global(w1 + (size_t)row * C_ + ks * 32 + half * 8);
      v16bf bm = b_from_lds(fB + l16 * C_ + ks * 32 + half * 8);
      acc = wmma_bf16(a, bm, acc);
    }
    const int o0 = mt * 16 + half * 8;       // D elem j -> o = o0 + j, n = l16
    const f32x4 g0 = ((const f32x4*)(bn1_g + o0))[0];
    const f32x4 g1 = ((const f32x4*)(bn1_g + o0))[1];
    const f32x4 vv0 = ((const f32x4*)(bn1_v + o0))[0];
    const f32x4 vv1 = ((const f32x4*)(bn1_v + o0))[1];
    const f32x4 bb0 = ((const f32x4*)(bn1_b + o0))[0];
    const f32x4 bb1 = ((const f32x4*)(bn1_b + o0))[1];
    const f32x4 mm0 = ((const f32x4*)(bn1_m + o0))[0];
    const f32x4 mm1 = ((const f32x4*)(bn1_m + o0))[1];
    unsigned short bv[8];
#pragma unroll
    for (int j = 0; j < 8; ++j) {
      const float g = (j < 4) ? g0[j & 3] : g1[j & 3];
      const float v = (j < 4) ? vv0[j & 3] : vv1[j & 3];
      const float bb = (j < 4) ? bb0[j & 3] : bb1[j & 3];
      const float m = (j < 4) ? mm0[j & 3] : mm1[j & 3];
      const float inv = g * RSQ(v + EPS_);
      bv[j] = f2bf(fmaxf(acc[j] * inv + (bb - m * inv), 0.f));
    }
    // scatter; mt<64 <=> o<1024 for every element of this tile (uniform branch)
    if (mt < 64) {                           // param1: o = c*CR + r
#pragma unroll
      for (int j = 0; j < 8; ++j) {
        const int o = o0 + j, cc = o >> 4, r = o & 15;
        p1T[l16 * (CR_ * C_) + r * C_ + cc] = bv[j];
      }
    } else {                                 // param2: o-1024 = r*C + c
#pragma unroll
      for (int j = 0; j < 8; ++j) {
        const int oo = o0 + j - C_ * CR_, r = oo >> 6, cc = oo & 63;
        p2T[l16 * (C_ * CR_) + cc * CR_ + r] = bv[j];
      }
    }
  }
  __syncthreads();

  // ---------------- phase 2: per-point matmuls + LayerNorms ------------------
#pragma unroll 1
  for (int rep = 0; rep < 2; ++rep) {
    const int nl = wv + rep * 8;             // local n column 0..15
    const int n  = n0 + nl;

    // (a) stage pre[b,:,n,:] -> preA[s][c] bf16 (coalesced over s, packed b32)
    const float* pb = pre + ((size_t)b * C_) * (N_ * S_) + (size_t)n * S_;
    unsigned int* preA32 = (unsigned int*)preA;
#pragma unroll 4
    for (int c = 0; c < C_; c += 2) {
      const float va = pb[(size_t)c * (N_ * S_) + lane];
      const float vb = pb[(size_t)(c + 1) * (N_ * S_) + lane];
      preA32[(lane * C_ + c) >> 1] = pk2(va, vb);
    }
    WB();

    // (b) t1[32x16] = preA[32x64] @ param1[64x16]
#pragma unroll
    for (int smt = 0; smt < 2; ++smt) {
      v8f acc = {};
      const int row = smt * 16 + l16;
#pragma unroll
      for (int ks = 0; ks < 2; ++ks) {
        v16bf a  = a_from_lds(preA + row * C_ + ks * 32 + half * 8);
        v16bf bm = b_from_lds(p1T + nl * (CR_ * C_) + l16 * C_ + ks * 32 + half * 8);
        acc = wmma_bf16(a, bm, acc);
      }
#pragma unroll
      for (int j = 0; j < 8; ++j)
        t1raw[(smt * 16 + j + half * 8) * CR_ + l16] = acc[j];
    }
    WB();

    // (c) LN over r (one s-row per lane) + relu -> t1A bf16 (packed b32)
    {
      float v[CR_]; float mu = 0.f;
#pragma unroll
      for (int r = 0; r < CR_; ++r) { v[r] = t1raw[lane * CR_ + r]; mu += v[r]; }
      mu *= (1.f / 16.f);
      float var = 0.f;
#pragma unroll
      for (int r = 0; r < CR_; ++r) { const float d = v[r] - mu; var += d * d; }
      const float rinv = RSQ(var * (1.f / 16.f) + EPS_);
      unsigned int* t1A32 = (unsigned int*)t1A;
#pragma unroll
      for (int r = 0; r < CR_; r += 2) {
        const float y0 = fmaxf((v[r]     - mu) * rinv * ln1_g[r]     + ln1_b[r],     0.f);
        const float y1 = fmaxf((v[r + 1] - mu) * rinv * ln1_g[r + 1] + ln1_b[r + 1], 0.f);
        t1A32[(lane * CR_ + r) >> 1] = pk2(y0, y1);
      }
    }
    WB();

    // (d) t2[32x64] = t1pad[32x32] @ param2pad[32x64]  (K=16 zero-padded)
    v16bf a2[2];
#pragma unroll
    for (int smt = 0; smt < 2; ++smt) {
      const unsigned short* p = t1A + (smt * 16 + l16) * CR_ + half * 8;
      u16x8 lo = *(const u16x8*)p;
      u16x16 t = {};
#pragma unroll
      for (int e = 0; e < 8; ++e) t[e] = lo[e];   // K = 8h..8h+7 (<16); K>=16 zero
      a2[smt] = __builtin_bit_cast(v16bf, t);
    }
    WB();   // a2 fully loaded before scratch reuse below

    v8f t2acc[2][4];
#pragma unroll
    for (int cnt = 0; cnt < 4; ++cnt) {
      v16bf bm;
      {
        u16x8 lo = *(const u16x8*)(p2T + nl * (C_ * CR_) + (cnt * 16 + l16) * CR_ + half * 8);
        u16x16 tb = {};
#pragma unroll
        for (int e = 0; e < 8; ++e) tb[e] = lo[e];  // K = 8h..8h+7 (<16); K>=16 zero
        bm = __builtin_bit_cast(v16bf, tb);
      }
#pragma unroll
      for (int smt = 0; smt < 2; ++smt) {
        v8f z = {};
        t2acc[smt][cnt] = wmma_bf16(a2[smt], bm, z);
      }
    }
#pragma unroll
    for (int smt = 0; smt < 2; ++smt)
#pragma unroll
      for (int cnt = 0; cnt < 4; ++cnt)
#pragma unroll
        for (int j = 0; j < 8; ++j)
          t2raw[(smt * 16 + j + half * 8) * C_ + cnt * 16 + l16] = t2acc[smt][cnt][j];
    WB();

    // (e) LN over c (one s-row per lane) + relu -> flat^T[nl][k=s*64+c] (b32)
    {
      float v[C_]; float mu = 0.f;
#pragma unroll
      for (int c = 0; c < C_; ++c) { v[c] = t2raw[lane * C_ + c]; mu += v[c]; }
      mu *= (1.f / 64.f);
      float var = 0.f;
#pragma unroll
      for (int c = 0; c < C_; ++c) { const float d = v[c] - mu; var += d * d; }
      const float rinv = RSQ(var * (1.f / 64.f) + EPS_);
      unsigned int* fl32 = (unsigned int*)flatT;
#pragma unroll
      for (int c = 0; c < C_; c += 2) {
        const float y0 = fmaxf((v[c]     - mu) * rinv * ln2_g[c]     + ln2_b[c],     0.f);
        const float y1 = fmaxf((v[c + 1] - mu) * rinv * ln2_g[c + 1] + ln2_b[c + 1], 0.f);
        fl32[(nl * (C_ * S_) + lane * C_ + c) >> 1] = pk2(y0, y1);
      }
    }
    WB();
  }
  __syncthreads();

  // ---------------- phase 3: y = w2 @ flat  (M=64, N=16, K=2048) -------------
  {
    const int mt = wv & 3;           // M-tile
    const int kh = wv >> 2;          // k-half: 2 waves per M-tile
    const int rowA = mt * 16 + l16;
    const float* w2row = w2 + (size_t)rowA * (C_ * S_);
    v8f acc = {};
#pragma unroll 4
    for (int i = 0; i < 32; ++i) {
      const int K0 = (kh * 32 + i) * 32;
      __builtin_prefetch(w2row + K0 + 64, 0, 0);           // global_prefetch_b8
      v16bf a  = a_from_global(w2row + K0 + half * 8);
      v16bf bm = b_from_lds(flatT + l16 * (C_ * S_) + K0 + half * 8);
      acc = wmma_bf16(a, bm, acc);
    }
#pragma unroll
    for (int j = 0; j < 8; ++j)
      atomicAdd(&yacc[(mt * 16 + j + half * 8) * NT_ + l16], acc[j]);  // ds_add_f32
  }
  __syncthreads();

  // ---------------- phase 4: BN2 + relu + residual + relu -> out -------------
#pragma unroll
  for (int q = 0; q < 4; ++q) {
    const int idx = tid + q * 256;           // 0..1023 = o*16 + j
    const int o = idx >> 4, j = idx & 15;
    const float inv = bn2_g[o] * RSQ(bn2_v[o] + EPS_);
    const float y = fmaxf(yacc[idx] * inv + (bn2_b[o] - bn2_m[o] * inv), 0.f);
    const size_t gi = ((size_t)(b * C_ + o)) * N_ + (n0 + j);
    out[gi] = fmaxf(y + feat[gi], 0.f);
  }
}

extern "C" void kernel_launch(void* const* d_in, const int* in_sizes, int n_in,
                              void* d_out, int out_size, void* d_ws, size_t ws_size,
                              hipStream_t stream) {
  (void)in_sizes; (void)n_in; (void)out_size; (void)d_ws; (void)ws_size;
  const float* pre   = (const float*)d_in[0];
  const float* feat  = (const float*)d_in[1];
  const float* w1    = (const float*)d_in[2];
  const float* bn1_g = (const float*)d_in[3];
  const float* bn1_b = (const float*)d_in[4];
  const float* bn1_m = (const float*)d_in[5];
  const float* bn1_v = (const float*)d_in[6];
  const float* ln1_g = (const float*)d_in[7];
  const float* ln1_b = (const float*)d_in[8];
  const float* ln2_g = (const float*)d_in[9];
  const float* ln2_b = (const float*)d_in[10];
  const float* w2    = (const float*)d_in[11];
  const float* bn2_g = (const float*)d_in[12];
  const float* bn2_b = (const float*)d_in[13];
  const float* bn2_m = (const float*)d_in[14];
  const float* bn2_v = (const float*)d_in[15];
  float* out = (float*)d_out;

  hipFuncSetAttribute(reinterpret_cast<const void*>(dyninter_fused),
                      hipFuncAttributeMaxDynamicSharedMemorySize, LDS_TOTAL);

  dim3 grid(B_ * (N_ / NT_));   // 2048 workgroups
  dim3 block(256);              // 8 wave32
  hipLaunchKernelGGL(dyninter_fused, grid, block, LDS_TOTAL, stream,
                     pre, feat, w1, bn1_g, bn1_b, bn1_m, bn1_v,
                     ln1_g, ln1_b, ln2_g, ln2_b, w2,
                     bn2_g, bn2_b, bn2_m, bn2_v, out);
}
